// ADDSEDiT_35459249996356
// MI455X (gfx1250) — matrix operand order repaired and checked by reading, not compiled, not run on hardware
//
#include <hip/hip_runtime.h>

typedef __attribute__((ext_vector_type(8)))  float  v8f;
typedef __attribute__((ext_vector_type(16))) __bf16 v16bf;

union Frag {
    v16bf v;
    uint4 u[2];
};

#define DIM   1024
#define SEQ   1024
#define BATCH 4
#define NHEAD 16
#define HDIM  64
#define SKIPF 0.70710678118654752f

// ---------------------------------------------------------------------------
// DPP lane-xor within 16-lane rows: xor1/2 via quad_perm, 4/8 via row_xmask.
// old=0 + bound_ctrl=1 so GCNDPPCombine can fold the DPP into the consumer.
// ---------------------------------------------------------------------------
template <int CTRL>
__device__ __forceinline__ float dpp_xor(float x) {
    int r = __builtin_amdgcn_update_dpp(0, __float_as_int(x), CTRL, 0xf, 0xf, true);
    return __int_as_float(r);
}
__device__ __forceinline__ float row_max16(float x) {
    x = fmaxf(x, dpp_xor<0xB1>(x));    // quad_perm xor 1
    x = fmaxf(x, dpp_xor<0x4E>(x));    // quad_perm xor 2
    x = fmaxf(x, dpp_xor<0x164>(x));   // row_xmask 4
    x = fmaxf(x, dpp_xor<0x168>(x));   // row_xmask 8
    return x;
}
__device__ __forceinline__ float row_sum16(float x) {
    x += dpp_xor<0xB1>(x);
    x += dpp_xor<0x4E>(x);
    x += dpp_xor<0x164>(x);
    x += dpp_xor<0x168>(x);
    return x;
}

// ---------------------------------------------------------------------------
// prep: s[b*D+k] = silu((c + t) * SKIP)
// ---------------------------------------------------------------------------
__global__ __launch_bounds__(256) void prep_kernel(const float* __restrict__ c,
                                                   const float* __restrict__ t,
                                                   float* __restrict__ s) {
    int i = blockIdx.x * 256 + threadIdx.x;   // B*D = 4096
    float v = (c[i] + t[i]) * SKIPF;
    s[i] = v / (1.f + __expf(-v));
}

// ---------------------------------------------------------------------------
// adaLN modulation: mod[b][j] = sum_k s[b][k] * w[k][j] + bias[j]
// ---------------------------------------------------------------------------
__global__ __launch_bounds__(256) void adaln_kernel(const float* __restrict__ s,
                                                    const float* __restrict__ w,
                                                    const float* __restrict__ bias,
                                                    float* __restrict__ mod) {
    int j = blockIdx.x * 256 + threadIdx.x;   // 6144
    int b = blockIdx.y;
    const float* sb = s + b * DIM;
    float acc = bias[j];
    for (int k = 0; k < DIM; ++k)
        acc += sb[k] * w[(size_t)k * (6 * DIM) + j];
    mod[(size_t)b * (6 * DIM) + j] = acc;
}

// ---------------------------------------------------------------------------
// Weight convert + transpose: Wt[n*K + k] = (bf16) W[k*N + n]
// ---------------------------------------------------------------------------
__global__ __launch_bounds__(256) void wcvt_kernel(const float* __restrict__ W,
                                                   __bf16* __restrict__ Wt,
                                                   int K, int N) {
    __shared__ __bf16 tile[32][33];
    int k0 = blockIdx.x * 32, n0 = blockIdx.y * 32;
    int t = threadIdx.x;
    {
        int tn = t & 31, tk = t >> 5;
#pragma unroll
        for (int p = 0; p < 4; ++p) {
            int k = tk + p * 8;
            tile[tn][k] = (__bf16)W[(size_t)(k0 + k) * N + n0 + tn];
        }
    }
    __syncthreads();
    {
        int tk = t & 31, tn = t >> 5;
#pragma unroll
        for (int p = 0; p < 4; ++p) {
            int n = tn + p * 8;
            Wt[(size_t)(n0 + n) * K + k0 + tk] = tile[n][tk];
        }
    }
}

// ---------------------------------------------------------------------------
// LayerNorm + modulate -> bf16 activation buffer.  One block per row.
// ---------------------------------------------------------------------------
__global__ __launch_bounds__(256) void ln_mod_kernel(const float* __restrict__ X,
                                                     const float* __restrict__ mod,
                                                     int shoff,
                                                     __bf16* __restrict__ H) {
    int row = blockIdx.x;           // B*L
    int b   = row >> 10;
    const float* x = X + (size_t)row * DIM;
    __shared__ float red[16];
    int t = threadIdx.x;

    float v[4];
    float s = 0.f, s2 = 0.f;
#pragma unroll
    for (int i = 0; i < 4; ++i) {
        v[i] = x[t + 256 * i];
        s  += v[i];
        s2 += v[i] * v[i];
    }
#pragma unroll
    for (int off = 1; off < 32; off <<= 1) {
        s  += __shfl_xor(s,  off, 32);
        s2 += __shfl_xor(s2, off, 32);
    }
    int w = t >> 5;
    if ((t & 31) == 0) { red[w] = s; red[8 + w] = s2; }
    __syncthreads();
    s = 0.f; s2 = 0.f;
#pragma unroll
    for (int i = 0; i < 8; ++i) { s += red[i]; s2 += red[8 + i]; }
    float mean = s * (1.f / DIM);
    float var  = s2 * (1.f / DIM) - mean * mean;
    float inv  = rsqrtf(var + 1e-6f);

    const float* shp = mod + (size_t)b * (6 * DIM) + shoff;
#pragma unroll
    for (int i = 0; i < 4; ++i) {
        int j = t + 256 * i;
        float sh = shp[j];
        float sc = shp[DIM + j];
        float hv = ((v[i] - mean) * inv * (1.f + sc) + sh) * SKIPF;
        H[(size_t)row * DIM + j] = (__bf16)hv;
    }
}

// ---------------------------------------------------------------------------
// WMMA GEMM: C[M,N] = A_bf16[M,K] @ Wt_bf16[N,K]^T + bias
// 128x128 block tile, 8 waves as 4(M) x 2(N), double-buffered LDS pipeline.
// EPI==0 : store f32.  EPI==1 : tanh-GELU then store bf16.
// ---------------------------------------------------------------------------
template <int EPI>
__global__ __launch_bounds__(256) void gemm_kernel(const __bf16* __restrict__ A,
                                                   const __bf16* __restrict__ Wt,
                                                   const float*  __restrict__ bias,
                                                   void* __restrict__ Cout,
                                                   int K, int N) {
    __shared__ __align__(16) __bf16 As[2][128 * 40];
    __shared__ __align__(16) __bf16 Bs[2][128 * 40];

    int t    = threadIdx.x;
    int w    = t >> 5, lane = t & 31;
    int lm   = lane & 15, kh = lane >> 4;
    int mw   = w >> 1, nw = w & 1;
    int row0 = blockIdx.y * 128;
    int col0 = blockIdx.x * 128;

    v8f zero = {0.f, 0.f, 0.f, 0.f, 0.f, 0.f, 0.f, 0.f};
    v8f acc[2][4];
#pragma unroll
    for (int i = 0; i < 2; ++i)
#pragma unroll
        for (int j = 0; j < 4; ++j) acc[i][j] = zero;

    int r = t >> 1, half = t & 1;
    const __bf16* arow = A  + (size_t)(row0 + r) * K + half * 16;
    const __bf16* brow = Wt + (size_t)(col0 + r) * K + half * 16;
    int soff = r * 40 + half * 16;

    {   // prologue: stage k-step 0 into buffer 0
        const uint4* sa = (const uint4*)(arow);
        const uint4* sb = (const uint4*)(brow);
        uint4* da = (uint4*)(&As[0][soff]);
        uint4* db = (uint4*)(&Bs[0][soff]);
        da[0] = sa[0]; da[1] = sa[1];
        db[0] = sb[0]; db[1] = sb[1];
    }
    __syncthreads();

    int nsteps = K >> 5;
    int cur = 0;
    for (int s = 0; s < nsteps; ++s) {
        uint4 na0, na1, nb0, nb1;
        bool more = (s + 1 < nsteps);
        if (more) {   // issue next tile's global loads
            const uint4* sa = (const uint4*)(arow + (s + 1) * 32);
            const uint4* sb = (const uint4*)(brow + (s + 1) * 32);
            na0 = sa[0]; na1 = sa[1];
            nb0 = sb[0]; nb1 = sb[1];
        }

        Frag af[2];
#pragma unroll
        for (int mi = 0; mi < 2; ++mi) {
            const __bf16* ar = &As[cur][(32 * mw + 16 * mi + lm) * 40];
            af[mi].u[0] = *(const uint4*)(ar + 8 * kh);
            af[mi].u[1] = *(const uint4*)(ar + 16 + 8 * kh);
        }
#pragma unroll
        for (int ni = 0; ni < 4; ++ni) {
            Frag bfr;
            const __bf16* br = &Bs[cur][(64 * nw + 16 * ni + lm) * 40];
            bfr.u[0] = *(const uint4*)(br + 8 * kh);
            bfr.u[1] = *(const uint4*)(br + 16 + 8 * kh);
#pragma unroll
            for (int mi = 0; mi < 2; ++mi)
                acc[mi][ni] = __builtin_amdgcn_wmma_f32_16x16x32_bf16(
                    false, af[mi].v, false, bfr.v, (short)0, acc[mi][ni], false, false);
        }

        if (more) {   // dump prefetched tile into other buffer
            uint4* da = (uint4*)(&As[cur ^ 1][soff]);
            uint4* db = (uint4*)(&Bs[cur ^ 1][soff]);
            da[0] = na0; da[1] = na1;
            db[0] = nb0; db[1] = nb1;
        }
        __syncthreads();
        cur ^= 1;
    }

#pragma unroll
    for (int mi = 0; mi < 2; ++mi)
#pragma unroll
        for (int ni = 0; ni < 4; ++ni) {
            int col = col0 + 64 * nw + 16 * ni + lm;
            float bv = bias[col];
#pragma unroll
            for (int rr = 0; rr < 8; ++rr) {
                int row = row0 + 32 * mw + 16 * mi + rr + 8 * kh;
                float v = acc[mi][ni][rr] + bv;
                if (EPI == 1) {
                    float u = 0.7978845608f * (v + 0.044715f * v * v * v);
                    v = 0.5f * v * (1.f + tanhf(u));
                    ((__bf16*)Cout)[(size_t)row * N + col] = (__bf16)v;
                } else {
                    ((float*)Cout)[(size_t)row * N + col] = v;
                }
            }
        }
}

// ---------------------------------------------------------------------------
// RoPE + split qkv into bf16 Q (prescaled by SCALE=0.125), K, V  [B,H,L,HD]
// ---------------------------------------------------------------------------
__global__ __launch_bounds__(256) void rope_kernel(const float* __restrict__ qkv,
                                                   __bf16* __restrict__ Qb,
                                                   __bf16* __restrict__ Kb,
                                                   __bf16* __restrict__ Vb) {
    size_t tid = (size_t)blockIdx.x * 256 + threadIdx.x;   // B*H*L*32
    int i = tid & 31;
    int l = (tid >> 5) & 1023;
    int h = (tid >> 15) & 15;
    int b = (int)(tid >> 19);

    const float* base = qkv + ((size_t)(b * SEQ + l)) * (3 * DIM) + h * HDIM + 2 * i;
    float omega = __powf(10000.f, -(float)(2 * i) / (float)HDIM);
    float ang = (float)l * omega;
    float cs = __cosf(ang), sn = __sinf(ang);

    size_t o = (((size_t)(b * NHEAD + h)) * SEQ + l) * HDIM + 2 * i;
    float q0 = base[0], q1 = base[1];
    Qb[o]     = (__bf16)((q0 * cs - q1 * sn) * 0.125f);
    Qb[o + 1] = (__bf16)((q1 * cs + q0 * sn) * 0.125f);
    float k0 = base[DIM], k1 = base[DIM + 1];
    Kb[o]     = (__bf16)(k0 * cs - k1 * sn);
    Kb[o + 1] = (__bf16)(k1 * cs + k0 * sn);
    Vb[o]     = (__bf16)base[2 * DIM];
    Vb[o + 1] = (__bf16)base[2 * DIM + 1];
}

// ---------------------------------------------------------------------------
// Flash attention: one block per (b,h,128 query rows); wave owns 16 rows.
// Double-buffered V tile; P relayout via per-wave LDS (same-wave, in-order).
// ---------------------------------------------------------------------------
__global__ __launch_bounds__(256) void attn_kernel(const __bf16* __restrict__ Q,
                                                   const __bf16* __restrict__ K,
                                                   const __bf16* __restrict__ V,
                                                   __bf16* __restrict__ Aout) {
    int blk = blockIdx.x;          // B*H*8
    int qb  = blk & 7;
    int bh  = blk >> 3;
    int b   = bh >> 4, h = bh & 15;
    const __bf16* Qp = Q + ((size_t)bh * SEQ + qb * 128) * HDIM;
    const __bf16* Kp = K + (size_t)bh * SEQ * HDIM;
    const __bf16* Vp = V + (size_t)bh * SEQ * HDIM;

    __shared__ __align__(16) __bf16 Vt[2][64 * 40];     // V transposed [d][key]
    __shared__ __align__(16) __bf16 Pw[8 * 16 * 40];    // per-wave P scratch

    int t = threadIdx.x;
    int w = t >> 5, lane = t & 31;
    int lm = lane & 15, kh = lane >> 4;

    Frag qa[2];
    {
        const __bf16* qrow = Qp + (16 * w + lm) * HDIM;
#pragma unroll
        for (int kk = 0; kk < 2; ++kk) {
            qa[kk].u[0] = *(const uint4*)(qrow + kk * 32 + 8 * kh);
            qa[kk].u[1] = *(const uint4*)(qrow + kk * 32 + 16 + 8 * kh);
        }
    }

    v8f zero = {0.f, 0.f, 0.f, 0.f, 0.f, 0.f, 0.f, 0.f};
    v8f acc[4];
    float mrow[8], lrow[8];
#pragma unroll
    for (int f = 0; f < 4; ++f) acc[f] = zero;
#pragma unroll
    for (int r = 0; r < 8; ++r) { mrow[r] = -1e30f; lrow[r] = 0.f; }

    union U8 { uint4 u; __bf16 e[8]; };
    int key = t >> 3, dbase = (t & 7) * 8;
    const __bf16* vsrc = Vp + (size_t)key * HDIM + dbase;

    {   // prologue: stage V tile 0
        U8 tmp;
        tmp.u = *(const uint4*)(vsrc);
#pragma unroll
        for (int e = 0; e < 8; ++e) Vt[0][(dbase + e) * 40 + key] = tmp.e[e];
    }
    __syncthreads();

    int cur = 0;
    for (int kt = 0; kt < 32; ++kt) {
        U8 tmpn;
        bool more = (kt < 31);
        if (more)   // issue next V tile's global load
            tmpn.u = *(const uint4*)(vsrc + (size_t)(kt + 1) * 32 * HDIM);

        // S = (SCALE*Q) K^T  for 32 keys -> two 16x16 fragments
        v8f S[2];
#pragma unroll
        for (int nk = 0; nk < 2; ++nk) {
            v8f s = zero;
#pragma unroll
            for (int kk = 0; kk < 2; ++kk) {
                Frag kb;
                const __bf16* krow = Kp + (size_t)(kt * 32 + nk * 16 + lm) * HDIM + kk * 32;
                kb.u[0] = *(const uint4*)(krow + 8 * kh);
                kb.u[1] = *(const uint4*)(krow + 16 + 8 * kh);
                s = __builtin_amdgcn_wmma_f32_16x16x32_bf16(
                        false, qa[kk].v, false, kb.v, (short)0, s, false, false);
            }
            S[nk] = s;
        }

        // online softmax via DPP row reductions
        float P0[8], P1[8];
#pragma unroll
        for (int r = 0; r < 8; ++r) {
            float mx    = row_max16(fmaxf(S[0][r], S[1][r]));
            float mnew  = fmaxf(mrow[r], mx);
            float alpha = __expf(mrow[r] - mnew);
            mrow[r] = mnew;
            float p0 = __expf(S[0][r] - mnew);
            float p1 = __expf(S[1][r] - mnew);
            P0[r] = p0; P1[r] = p1;
            float sum = row_sum16(p0 + p1);
            lrow[r] = lrow[r] * alpha + sum;
#pragma unroll
            for (int f = 0; f < 4; ++f) acc[f][r] *= alpha;
        }

        // P: C-layout -> A-layout via per-wave LDS round trip
        __bf16* pw = Pw + w * 16 * 40;
#pragma unroll
        for (int r = 0; r < 8; ++r) {
            int row = r + 8 * kh;
            pw[row * 40 + lm]      = (__bf16)P0[r];
            pw[row * 40 + 16 + lm] = (__bf16)P1[r];
        }
        asm volatile("s_wait_dscnt 0" ::: "memory");
        Frag pa;
        pa.u[0] = *(const uint4*)(pw + lm * 40 + 8 * kh);
        pa.u[1] = *(const uint4*)(pw + lm * 40 + 16 + 8 * kh);

        // acc += P @ V
#pragma unroll
        for (int f = 0; f < 4; ++f) {
            Frag vb;
            const __bf16* vrow = &Vt[cur][(f * 16 + lm) * 40];
            vb.u[0] = *(const uint4*)(vrow + 8 * kh);
            vb.u[1] = *(const uint4*)(vrow + 16 + 8 * kh);
            acc[f] = __builtin_amdgcn_wmma_f32_16x16x32_bf16(
                         false, pa.v, false, vb.v, (short)0, acc[f], false, false);
        }

        if (more) {   // dump prefetched V tile into other buffer
#pragma unroll
            for (int e = 0; e < 8; ++e)
                Vt[cur ^ 1][(dbase + e) * 40 + key] = tmpn.e[e];
        }
        __syncthreads();
        cur ^= 1;
    }

#pragma unroll
    for (int r = 0; r < 8; ++r) {
        float inv = 1.f / lrow[r];
        int row = qb * 128 + 16 * w + r + 8 * kh;
        size_t orow = ((size_t)b * SEQ + row) * DIM + h * HDIM;
#pragma unroll
        for (int f = 0; f < 4; ++f)
            Aout[orow + f * 16 + lm] = (__bf16)(acc[f][r] * inv);
    }
}

// ---------------------------------------------------------------------------
// gated residual: Xout = (X + g * O) * SKIP,  g = mod[b][goff + d]
// ---------------------------------------------------------------------------
__global__ __launch_bounds__(256) void resid_kernel(const float* __restrict__ X,
                                                    const float* __restrict__ O,
                                                    const float* __restrict__ mod,
                                                    int goff,
                                                    float* __restrict__ Xout) {
    size_t idx = (size_t)blockIdx.x * 256 + threadIdx.x;   // B*L*D = 2^22
    int d = idx & 1023;
    int b = (int)(idx >> 20);
    float g = mod[(size_t)b * (6 * DIM) + goff + d];
    Xout[idx] = (X[idx] + g * O[idx]) * SKIPF;
}

// ---------------------------------------------------------------------------
extern "C" void kernel_launch(void* const* d_in, const int* in_sizes, int n_in,
                              void* d_out, int out_size, void* d_ws, size_t ws_size,
                              hipStream_t stream) {
    const float* x_in    = (const float*)d_in[0];
    const float* c_in    = (const float*)d_in[1];
    const float* t_in    = (const float*)d_in[2];
    const float* qkv_w   = (const float*)d_in[3];
    const float* qkv_b   = (const float*)d_in[4];
    const float* proj_w  = (const float*)d_in[5];
    const float* proj_b  = (const float*)d_in[6];
    const float* mlp_w1  = (const float*)d_in[7];
    const float* mlp_b1  = (const float*)d_in[8];
    const float* mlp_w2  = (const float*)d_in[9];
    const float* mlp_b2  = (const float*)d_in[10];
    const float* adaln_w = (const float*)d_in[11];
    const float* adaln_b = (const float*)d_in[12];

    char* ws = (char*)d_ws;
    size_t off = 0;
    auto alloc = [&](size_t bytes) {
        size_t o = off;
        off += (bytes + 255) & ~(size_t)255;
        return o;
    };
    const size_t BL = (size_t)BATCH * SEQ;                 // 4096 rows
    float*  sbuf = (float*)(ws + alloc(BATCH * DIM * 4));
    float*  mod  = (float*)(ws + alloc(BATCH * 6 * DIM * 4));
    __bf16* hbuf = (__bf16*)(ws + alloc(BL * DIM * 2));
    float*  qkv  = (float*)(ws + alloc(BL * 3 * DIM * 4));
    __bf16* Qb   = (__bf16*)(ws + alloc(BL * DIM * 2));
    __bf16* Kb   = (__bf16*)(ws + alloc(BL * DIM * 2));
    __bf16* Vb   = (__bf16*)(ws + alloc(BL * DIM * 2));
    __bf16* aout = (__bf16*)(ws + alloc(BL * DIM * 2));
    float*  obuf = (float*)(ws + alloc(BL * DIM * 4));
    __bf16* m1   = (__bf16*)(ws + alloc(BL * 4 * DIM * 2));
    float*  mbuf = (float*)(ws + alloc(BL * DIM * 4));
    float*  xws  = (float*)(ws + alloc(BL * DIM * 4));
    __bf16* wt   = (__bf16*)(ws + alloc((size_t)4 * DIM * DIM * 2));  // max N*K

    prep_kernel<<<16, 256, 0, stream>>>(c_in, t_in, sbuf);

    for (int i = 0; i < 4; ++i) {
        adaln_kernel<<<dim3(24, 4), 256, 0, stream>>>(
            sbuf, adaln_w + (size_t)i * DIM * 6 * DIM, adaln_b + (size_t)i * 6 * DIM, mod);

        const float* xc = (i == 0) ? x_in : xws;

        // --- attention branch ---
        ln_mod_kernel<<<4096, 256, 0, stream>>>(xc, mod, 0, hbuf);

        wcvt_kernel<<<dim3(DIM / 32, 3 * DIM / 32), 256, 0, stream>>>(
            qkv_w + (size_t)i * DIM * 3 * DIM, wt, DIM, 3 * DIM);
        gemm_kernel<0><<<dim3(24, 32), 256, 0, stream>>>(
            hbuf, wt, qkv_b + (size_t)i * 3 * DIM, qkv, DIM, 3 * DIM);

        rope_kernel<<<8192, 256, 0, stream>>>(qkv, Qb, Kb, Vb);
        attn_kernel<<<512, 256, 0, stream>>>(Qb, Kb, Vb, aout);

        wcvt_kernel<<<dim3(DIM / 32, DIM / 32), 256, 0, stream>>>(
            proj_w + (size_t)i * DIM * DIM, wt, DIM, DIM);
        gemm_kernel<0><<<dim3(8, 32), 256, 0, stream>>>(
            aout, wt, proj_b + (size_t)i * DIM, obuf, DIM, DIM);

        resid_kernel<<<16384, 256, 0, stream>>>(xc, obuf, mod, 2 * DIM, xws);

        // --- mlp branch ---
        ln_mod_kernel<<<4096, 256, 0, stream>>>(xws, mod, 3 * DIM, hbuf);

        wcvt_kernel<<<dim3(DIM / 32, 4 * DIM / 32), 256, 0, stream>>>(
            mlp_w1 + (size_t)i * DIM * 4 * DIM, wt, DIM, 4 * DIM);
        gemm_kernel<1><<<dim3(32, 32), 256, 0, stream>>>(
            hbuf, wt, mlp_b1 + (size_t)i * 4 * DIM, m1, DIM, 4 * DIM);

        wcvt_kernel<<<dim3(4 * DIM / 32, DIM / 32), 256, 0, stream>>>(
            mlp_w2 + (size_t)i * 4 * DIM * DIM, wt, 4 * DIM, DIM);
        gemm_kernel<0><<<dim3(8, 32), 256, 0, stream>>>(
            m1, wt, mlp_b2 + (size_t)i * DIM, mbuf, 4 * DIM, DIM);

        float* xo = (i == 3) ? (float*)d_out : xws;
        resid_kernel<<<16384, 256, 0, stream>>>(xws, mbuf, mod, 5 * DIM, xo);
    }
}
